// Model_12945031430925
// MI455X (gfx1250) — matrix operand order repaired and checked by reading, not compiled, not run on hardware
//
#include <hip/hip_runtime.h>
#include <hip/hip_bf16.h>

// ---------------------------------------------------------------------------
// VQ-VAE forward for MI455X (gfx1250, wave32).
// All convs run as implicit GEMM on v_wmma_f32_16x16x32_f16 (f32 accum).
// Activations stored f16 [b][t+4][C] (t-major -> contiguous K-dim fragments),
// weights f16 [k][co][ci]. ~585 GFLOP total, L2-resident working set.
// Each wave computes a 16(co) x 64(t) strip: A fragment reused across 4
// independent accumulator chains (better vmem/wmma ratio + WMMA ILP).
// ---------------------------------------------------------------------------

#define BB 8
#define TT 4096
#define TP (TT + 4)   // 2 zero rows of time padding on each side

typedef __attribute__((ext_vector_type(16))) _Float16 v16h;
typedef __attribute__((ext_vector_type(8)))  _Float16 v8h;
typedef __attribute__((ext_vector_type(8)))  float    v8f;
typedef __attribute__((ext_vector_type(4)))  float    v4f;

static inline int cdiv(int a, int b) { return (a + b - 1) / b; }

// ----------------------------- utility kernels -----------------------------

__global__ void k_zero_f32(float* p, int n) {
  int i = blockIdx.x * blockDim.x + threadIdx.x;
  if (i < n) p[i] = 0.f;
}

// zero the 4 time-padding rows (2 front, 2 back) of an f16 [B][TP][Cp] buffer
__global__ void k_pad_zero(_Float16* dst, int Cp) {
  int i = blockIdx.x * blockDim.x + threadIdx.x;
  int tot = BB * 4 * Cp;
  if (i >= tot) return;
  int c  = i % Cp;
  int r4 = (i / Cp) % 4;
  int b  = i / (4 * Cp);
  int row = (r4 < 2) ? r4 : (TP - 4 + r4);
  dst[((size_t)b * TP + row) * Cp + c] = (_Float16)0.f;
}

// weights -> f16 [k][co][ci_pad]; conv src [Co,Ci,K]; deconv src [Ci,Co,K] w/ k flip
__global__ void k_cvt_w(const float* __restrict__ src, _Float16* __restrict__ dst,
                        int Co, int Ci, int Cip, int Kk, int deconv) {
  int i = blockIdx.x * blockDim.x + threadIdx.x;
  int tot = Kk * Co * Cip;
  if (i >= tot) return;
  int c = i % Cip;
  int o = (i / Cip) % Co;
  int k = i / (Cip * Co);
  float v = 0.f;
  if (c < Ci)
    v = deconv ? src[((size_t)c * Co + o) * Kk + (Kk - 1 - k)]
               : src[((size_t)o * Ci + c) * Kk + k];
  dst[i] = (_Float16)v;
}

// x [B,80,T] f32 -> f16 [B][TP][96] (channels 80..95 zero padded)
__global__ void k_cvt_x0(const float* __restrict__ x, _Float16* __restrict__ dst) {
  int i = blockIdx.x * blockDim.x + threadIdx.x;
  int tot = BB * TT * 96;
  if (i >= tot) return;
  int c = i % 96;
  int t = (i / 96) % TT;
  int b = i / (96 * TT);
  float v = (c < 80) ? x[((size_t)b * 80 + c) * TT + t] : 0.f;
  dst[((size_t)b * TP + t + 2) * 96 + c] = (_Float16)v;
}

// L2-normalize codebook rows [512][64]
__global__ void k_cbnorm(const float* __restrict__ cb, float* __restrict__ cbn) {
  int nz = blockIdx.x * blockDim.x + threadIdx.x;
  if (nz >= 512) return;
  float s = 0.f;
  for (int j = 0; j < 64; ++j) { float v = cb[nz * 64 + j]; s += v * v; }
  float rs = rsqrtf(s + 1e-12f);
  for (int j = 0; j < 64; ++j) cbn[nz * 64 + j] = cb[nz * 64 + j] * rs;
}

// normalized speaker embedding lookup -> ye [B][128] f32
__global__ void k_spk(const float* __restrict__ spk, const int* __restrict__ y,
                      float* __restrict__ ye) {
  int b = threadIdx.x;
  if (b >= BB) return;
  int yi = y[b];   // y shape [B,1]
  float s = 0.f;
  for (int j = 0; j < 128; ++j) { float v = spk[yi * 128 + j]; s += v * v; }
  float rs = rsqrtf(s + 1e-12f);
  for (int j = 0; j < 128; ++j) ye[b * 128 + j] = spk[yi * 128 + j] * rs;
}

// broadcast cond ye into channels [coff, coff+128) of f16 [B][TP][Cp] buffer
__global__ void k_cond(const float* __restrict__ ye, _Float16* __restrict__ dst,
                       int Cp, int coff) {
  int i = blockIdx.x * blockDim.x + threadIdx.x;
  int tot = BB * TT * 128;
  if (i >= tot) return;
  int j = i % 128;
  int t = (i / 128) % TT;
  int b = i / (128 * TT);
  dst[((size_t)b * TP + t + 2) * Cp + coff + j] = (_Float16)ye[b * 128 + j];
}

// ------------------------- WMMA implicit-GEMM conv -------------------------
// in:  f16 [B][TP][Cin]   (Cin multiple of 32)
// w:   f16 [k][Co][Cin]
// out: mode 0/1 -> f32 [B][T][Co] (+ per-sample sum/sumsq atomics if mode==0)
//      mode 2   -> f32 [B][Co][T] (final output, bias only)
// One wave per 16(co) x 64(t) strip: shared A fragment, 4 independent
// accumulator chains (4 adjacent 16-wide t-subtiles).
__global__ __launch_bounds__(128) void k_conv_wmma(
    const _Float16* __restrict__ in, const _Float16* __restrict__ w,
    const float* __restrict__ bias, float* __restrict__ out,
    float* __restrict__ stats, int Co, int Cin, int Kk, int k0off, int mode) {
  const int lane = threadIdx.x & 31;
  const int wv   = threadIdx.x >> 5;
  const int cotile = blockIdx.y * 4 + wv;
  if (cotile * 16 >= Co) return;               // wave-uniform: EXEC stays all-1
  const int co0 = cotile * 16;
  const int t0  = blockIdx.x * 64;
  const int b   = blockIdx.z;
  const int n   = lane & 15;                   // A-row m == B-col n == lane&15
  const int hf  = lane >> 4;
  const size_t rstep = (size_t)16 * Cin;       // 16 time rows between subtiles

  v8f acc[4];
  acc[0] = {}; acc[1] = {}; acc[2] = {}; acc[3] = {};
  for (int k = 0; k < Kk; ++k) {
    const _Float16* wrow = w  + ((size_t)(k * Co + co0 + n)) * Cin;
    const _Float16* brow = in + ((size_t)b * TP + t0 + n + k + k0off) * Cin;
    for (int kk = 0; kk < Cin; kk += 32) {
      // A 16x32 f16 fragment: V0-7 hold K = hf*16+{0..15}
      v16h av;
      *(v8h*)&av         = *(const v8h*)(wrow + kk + hf * 16);
      *(((v8h*)&av) + 1) = *(const v8h*)(wrow + kk + hf * 16 + 8);
      // B 32x16 f16 fragments for 4 t-subtiles: V0-7 hold K = hf*16+{0..15}
      const _Float16* bp = brow + kk + hf * 16;
      v16h b0, b1, b2, b3;
      *(v8h*)&b0         = *(const v8h*)(bp);
      *(((v8h*)&b0) + 1) = *(const v8h*)(bp + 8);
      *(v8h*)&b1         = *(const v8h*)(bp + rstep);
      *(((v8h*)&b1) + 1) = *(const v8h*)(bp + rstep + 8);
      *(v8h*)&b2         = *(const v8h*)(bp + 2 * rstep);
      *(((v8h*)&b2) + 1) = *(const v8h*)(bp + 2 * rstep + 8);
      *(v8h*)&b3         = *(const v8h*)(bp + 3 * rstep);
      *(((v8h*)&b3) + 1) = *(const v8h*)(bp + 3 * rstep + 8);
      acc[0] = __builtin_amdgcn_wmma_f32_16x16x32_f16(false, av, false, b0,
                                                      (short)0, acc[0], false, false);
      acc[1] = __builtin_amdgcn_wmma_f32_16x16x32_f16(false, av, false, b1,
                                                      (short)0, acc[1], false, false);
      acc[2] = __builtin_amdgcn_wmma_f32_16x16x32_f16(false, av, false, b2,
                                                      (short)0, acc[2], false, false);
      acc[3] = __builtin_amdgcn_wmma_f32_16x16x32_f16(false, av, false, b3,
                                                      (short)0, acc[3], false, false);
    }
  }

  if (mode == 2) {  // final layer: write [B][Co][T] directly
#pragma unroll
    for (int j = 0; j < 4; ++j) {
#pragma unroll
      for (int r = 0; r < 8; ++r) {
        int co = co0 + 8 * hf + r;             // D tile: M = r + 8*(lane>=16)
        out[((size_t)b * Co + co) * TT + t0 + 16 * j + n] = acc[j][r] + bias[co];
      }
    }
    return;
  }

  float lsum = 0.f, lsq = 0.f;
#pragma unroll
  for (int j = 0; j < 4; ++j) {
    float vs[8];
#pragma unroll
    for (int r = 0; r < 8; ++r) {
      float v = acc[j][r] + bias[co0 + 8 * hf + r];
      vs[r] = v;
      lsum += v;
      lsq  += v * v;
    }
    float* dp = out + ((size_t)b * TT + t0 + 16 * j + n) * Co + co0 + 8 * hf;
    v4f p0 = {vs[0], vs[1], vs[2], vs[3]};
    v4f p1 = {vs[4], vs[5], vs[6], vs[7]};
    *(v4f*)dp       = p0;                      // 32B contiguous store
    *(v4f*)(dp + 4) = p1;
  }

  if (mode == 0) {                             // GroupNorm partial sums
#pragma unroll
    for (int off = 16; off > 0; off >>= 1) {
      lsum += __shfl_xor(lsum, off, 32);
      lsq  += __shfl_xor(lsq,  off, 32);
    }
    if (lane == 0) {
      atomicAdd(stats + 2 * b,     lsum);
      atomicAdd(stats + 2 * b + 1, lsq);
    }
  }
}

// ------------------------- GroupNorm epilogue passes -----------------------

__global__ void k_apply_lrelu(const float* __restrict__ Y, const float* __restrict__ st,
                              const float* __restrict__ g, const float* __restrict__ be,
                              _Float16* __restrict__ dst, int Co, int dstCp) {
  int i = blockIdx.x * blockDim.x + threadIdx.x;
  int tot = BB * TT * Co;
  if (i >= tot) return;
  int c = i % Co;
  int t = (i / Co) % TT;
  int b = i / (Co * TT);
  float inv = 1.f / ((float)Co * (float)TT);
  float mu  = st[2 * b] * inv;
  float var = st[2 * b + 1] * inv - mu * mu;
  float rs  = rsqrtf(var + 1e-5f);
  float v = (Y[((size_t)b * TT + t) * Co + c] - mu) * rs * g[c] + be[c];
  v = (v >= 0.f) ? v : 0.02f * v;
  dst[((size_t)b * TP + t + 2) * dstCp + c] = (_Float16)v;
}

__global__ void k_apply_glu(const float* __restrict__ Y, const float* __restrict__ st,
                            const float* __restrict__ g, const float* __restrict__ be,
                            _Float16* __restrict__ dst, int Cho, int dstCp) {
  int i = blockIdx.x * blockDim.x + threadIdx.x;
  int tot = BB * TT * Cho;
  if (i >= tot) return;
  int c = i % Cho;
  int t = (i / Cho) % TT;
  int b = i / (Cho * TT);
  int Ctot = 2 * Cho;
  float inv = 1.f / ((float)Ctot * (float)TT);
  float mu  = st[2 * b] * inv;
  float var = st[2 * b + 1] * inv - mu * mu;
  float rs  = rsqrtf(var + 1e-5f);
  const float* row = Y + ((size_t)b * TT + t) * Ctot;
  float a  = (row[c] - mu) * rs * g[c] + be[c];
  float gt = (row[c + Cho] - mu) * rs * g[c + Cho] + be[c + Cho];
  float v = a * (1.f / (1.f + expf(-gt)));
  dst[((size_t)b * TP + t + 2) * dstCp + c] = (_Float16)v;
}

// ----------------------------- vector quantizer ----------------------------
// argmin ||z^ - c^||^2 == argmax z . c^  (codebook pre-normalized).
__global__ __launch_bounds__(128) void k_vq(const float* __restrict__ Z,
                                            const float* __restrict__ cbn,
                                            _Float16* __restrict__ dst, int dstCp) {
  int i = blockIdx.x * blockDim.x + threadIdx.x;
  if (i >= BB * TT) return;
  int t = i % TT;
  int b = i / TT;
  const float* z = Z + (size_t)i * 64;
  float zr[64];
#pragma unroll
  for (int j = 0; j < 64; ++j) zr[j] = z[j];
  int best = 0;
  float bv = -3.4e38f;
  for (int nn = 0; nn < 512; ++nn) {
    const float* cr = cbn + nn * 64;
    float d = 0.f;
#pragma unroll
    for (int j = 0; j < 64; ++j) d += zr[j] * cr[j];
    if (d > bv) { bv = d; best = nn; }
  }
  const float* cr = cbn + best * 64;
  _Float16* o = dst + ((size_t)b * TP + t + 2) * dstCp;
#pragma unroll
  for (int j = 0; j < 64; ++j) o[j] = (_Float16)cr[j];
}

// --------------------------------- driver ----------------------------------

extern "C" void kernel_launch(void* const* d_in, const int* in_sizes, int n_in,
                              void* d_out, int out_size, void* d_ws, size_t ws_size,
                              hipStream_t stream) {
  (void)in_sizes; (void)n_in; (void)out_size; (void)ws_size;
  const float* x   = (const float*)d_in[0];
  const int*   y   = (const int*)d_in[1];
  const float* e0w = (const float*)d_in[2],  *e0b = (const float*)d_in[3];
  const float* e0g = (const float*)d_in[4],  *e0be = (const float*)d_in[5];
  const float* e1w = (const float*)d_in[6],  *e1b = (const float*)d_in[7];
  const float* e1g = (const float*)d_in[8],  *e1be = (const float*)d_in[9];
  const float* e2w = (const float*)d_in[10], *e2b = (const float*)d_in[11];
  const float* e2g = (const float*)d_in[12], *e2be = (const float*)d_in[13];
  const float* e3w = (const float*)d_in[14], *e3b = (const float*)d_in[15];
  const float* e3g = (const float*)d_in[16], *e3be = (const float*)d_in[17];
  const float* mw  = (const float*)d_in[18], *mb  = (const float*)d_in[19];
  const float* cb  = (const float*)d_in[20], *spk = (const float*)d_in[21];
  const float* d0w = (const float*)d_in[22], *d0b = (const float*)d_in[23];
  const float* d0g = (const float*)d_in[24], *d0be = (const float*)d_in[25];
  const float* d1w = (const float*)d_in[26], *d1b = (const float*)d_in[27];
  const float* d1g = (const float*)d_in[28], *d1be = (const float*)d_in[29];
  const float* d2w = (const float*)d_in[30], *d2b = (const float*)d_in[31];

  char* ws = (char*)d_ws;
  size_t off = 0;
  auto take = [&](size_t bytes) -> char* {
    char* p = ws + off;
    off = (off + bytes + 255) & ~(size_t)255;
    return p;
  };
  float*    Y   = (float*)take((size_t)BB * TT * 1024 * 4);     // conv f32 scratch
  _Float16* X0  = (_Float16*)take((size_t)BB * TP * 96 * 2);
  _Float16* HA  = (_Float16*)take((size_t)BB * TP * 512 * 2);
  _Float16* HB  = (_Float16*)take((size_t)BB * TP * 512 * 2);
  _Float16* D0  = (_Float16*)take((size_t)BB * TP * 192 * 2);
  _Float16* D1  = (_Float16*)take((size_t)BB * TP * 640 * 2);
  _Float16* D2  = (_Float16*)take((size_t)BB * TP * 640 * 2);
  _Float16* WE0 = (_Float16*)take((size_t)5 * 512 * 96 * 2);
  _Float16* WE1 = (_Float16*)take((size_t)5 * 512 * 512 * 2);
  _Float16* WE2 = (_Float16*)take((size_t)5 * 512 * 512 * 2);
  _Float16* WE3 = (_Float16*)take((size_t)5 * 512 * 512 * 2);
  _Float16* WM  = (_Float16*)take((size_t)1 * 64 * 512 * 2);
  _Float16* WD0 = (_Float16*)take((size_t)5 * 1024 * 192 * 2);
  _Float16* WD1 = (_Float16*)take((size_t)5 * 1024 * 640 * 2);
  _Float16* WD2 = (_Float16*)take((size_t)5 * 80 * 640 * 2);
  float*    CBN = (float*)take((size_t)512 * 64 * 4);
  float*    YE  = (float*)take((size_t)BB * 128 * 4);
  float*    ST  = (float*)take((size_t)12 * 4);                 // 6 GN layers x (sum,sumsq)

  auto conv = [&](const _Float16* in, const _Float16* w, const float* bias,
                  float* out, float* st, int Co, int Cin, int Kk, int k0off, int mode) {
    dim3 g(TT / 64, cdiv(Co / 16, 4), BB);
    k_conv_wmma<<<g, dim3(128), 0, stream>>>(in, w, bias, out, st, Co, Cin, Kk, k0off, mode);
  };
  auto padz = [&](_Float16* d, int Cp) {
    int tot = BB * 4 * Cp;
    k_pad_zero<<<cdiv(tot, 256), 256, 0, stream>>>(d, Cp);
  };
  auto cw = [&](const float* s, _Float16* d, int Co, int Ci, int Cip, int Kk, int dec) {
    int tot = Kk * Co * Cip;
    k_cvt_w<<<cdiv(tot, 256), 256, 0, stream>>>(s, d, Co, Ci, Cip, Kk, dec);
  };

  // init
  k_zero_f32<<<1, 32, 0, stream>>>(ST, 12);
  padz(X0, 96); padz(HA, 512); padz(HB, 512);
  padz(D0, 192); padz(D1, 640); padz(D2, 640);
  cw(e0w, WE0, 512, 80, 96, 5, 0);
  cw(e1w, WE1, 512, 512, 512, 5, 0);
  cw(e2w, WE2, 512, 512, 512, 5, 0);
  cw(e3w, WE3, 512, 512, 512, 5, 0);
  cw(mw,  WM,  64, 512, 512, 1, 0);
  cw(d0w, WD0, 1024, 192, 192, 5, 1);
  cw(d1w, WD1, 1024, 640, 640, 5, 1);
  cw(d2w, WD2, 80, 640, 640, 5, 1);
  k_cvt_x0<<<cdiv(BB * TT * 96, 256), 256, 0, stream>>>(x, X0);
  k_cbnorm<<<cdiv(512, 64), 64, 0, stream>>>(cb, CBN);
  k_spk<<<1, 32, 0, stream>>>(spk, y, YE);
  k_cond<<<cdiv(BB * TT * 128, 256), 256, 0, stream>>>(YE, D0, 192, 64);
  k_cond<<<cdiv(BB * TT * 128, 256), 256, 0, stream>>>(YE, D1, 640, 512);
  k_cond<<<cdiv(BB * TT * 128, 256), 256, 0, stream>>>(YE, D2, 640, 512);

  // encoder
  conv(X0, WE0, e0b, Y, ST + 0, 512, 96, 5, 0, 0);
  k_apply_lrelu<<<cdiv(BB * TT * 512, 256), 256, 0, stream>>>(Y, ST + 0, e0g, e0be, HA, 512, 512);
  conv(HA, WE1, e1b, Y, ST + 2, 512, 512, 5, 0, 0);
  k_apply_lrelu<<<cdiv(BB * TT * 512, 256), 256, 0, stream>>>(Y, ST + 2, e1g, e1be, HB, 512, 512);
  conv(HB, WE2, e2b, Y, ST + 4, 512, 512, 5, 0, 0);
  k_apply_lrelu<<<cdiv(BB * TT * 512, 256), 256, 0, stream>>>(Y, ST + 4, e2g, e2be, HA, 512, 512);
  conv(HA, WE3, e3b, Y, ST + 6, 512, 512, 5, 0, 0);
  k_apply_lrelu<<<cdiv(BB * TT * 512, 256), 256, 0, stream>>>(Y, ST + 6, e3g, e3be, HB, 512, 512);
  conv(HB, WM, mb, Y, nullptr, 64, 512, 1, 2, 1);             // z -> Y [b][t][64]

  // vector quantize -> decoder input channels 0..63 (cond already at 64..191)
  k_vq<<<cdiv(BB * TT, 128), 128, 0, stream>>>(Y, CBN, D0, 192);

  // decoder
  conv(D0, WD0, d0b, Y, ST + 8, 1024, 192, 5, 0, 0);
  k_apply_glu<<<cdiv(BB * TT * 512, 256), 256, 0, stream>>>(Y, ST + 8, d0g, d0be, D1, 512, 640);
  conv(D1, WD1, d1b, Y, ST + 10, 1024, 640, 5, 0, 0);
  k_apply_glu<<<cdiv(BB * TT * 512, 256), 256, 0, stream>>>(Y, ST + 10, d1g, d1be, D2, 512, 640);
  conv(D2, WD2, d2b, (float*)d_out, nullptr, 80, 640, 5, 0, 2);
}
